// GAT_13297218748807
// MI455X (gfx1250) — matrix operand order repaired and checked by reading, not compiled, not run on hardware
//
#include <hip/hip_runtime.h>

// ---------------------------------------------------------------------------
// GAT on MI455X (gfx1250): mask-bitset + bf16-WMMA flash-style attention,
// async double-buffered LDS staging (GLOBAL_LOAD_ASYNC_TO_LDS_* + ASYNCcnt).
// ---------------------------------------------------------------------------

constexpr int kN    = 8192;
constexpr int kFIN  = 256;
constexpr int kHID  = 64;
constexpr int kNH   = 4;
constexpr int kNCLS = 16;
constexpr int kMW   = kN / 32;          // mask words per row (256)

typedef __attribute__((ext_vector_type(16))) __bf16 v16bf;
typedef __attribute__((ext_vector_type(8)))  float  v8f;

union BF16Frag {
    v16bf v;
    unsigned short u[16];
    unsigned int   w[8];
};

// Hardware bf16 conversion (v_cvt_*_bf16_f32, RNE) instead of a 4-op
// software round-to-nearest-even sequence.
static __device__ __forceinline__ unsigned int packbf(float a, float b) {
    union { __bf16 h[2]; unsigned int w; } u;
    u.h[0] = (__bf16)a;
    u.h[1] = (__bf16)b;
    return u.w;
}
static __device__ __forceinline__ unsigned short bf16bits(float a) {
    union { __bf16 h; unsigned short s; } u;
    u.h = (__bf16)a;
    return u.s;
}
static __device__ __forceinline__ float bf2f(unsigned int hbits) {
    return __uint_as_float(hbits << 16);
}
static __device__ __forceinline__ unsigned int ballot32(bool p) {
#if __has_builtin(__builtin_amdgcn_ballot_w32)
    return __builtin_amdgcn_ballot_w32(p);
#else
    return (unsigned int)__ballot(p);
#endif
}

#define HAVE_ASYNC_LDS (__has_builtin(__builtin_amdgcn_global_load_async_to_lds_b128) && \
                        __has_builtin(__builtin_amdgcn_global_load_async_to_lds_b32) &&  \
                        __has_builtin(__builtin_amdgcn_s_wait_asynccnt))

#if HAVE_ASYNC_LDS
typedef int v4i_vs __attribute__((vector_size(16)));
// Parameter types per hipcc diagnostics:
//   b128: (__device__ int __vector(4)*, __shared__ ..., Imm offset, Imm cpol)
//   b32 : (__device__ int*,             __shared__ ..., Imm offset, Imm cpol)
static __device__ __forceinline__ void async_b128(const void* g, void* l) {
    __builtin_amdgcn_global_load_async_to_lds_b128(
        (__attribute__((address_space(1))) v4i_vs*)g,
        (__attribute__((address_space(3))) v4i_vs*)l, 0, 0);
}
static __device__ __forceinline__ void async_b32(const void* g, void* l) {
    __builtin_amdgcn_global_load_async_to_lds_b32(
        (__attribute__((address_space(1))) int*)g,
        (__attribute__((address_space(3))) int*)l, 0, 0);
}
#endif

// --------------------------------------------------------------------------
// K0: bias_mat (fp32, 256 MB) -> adjacency bitset (8 MB, L2-resident after).
// --------------------------------------------------------------------------
__global__ __launch_bounds__(256) void k_build_mask(const float* __restrict__ bias,
                                                    unsigned int* __restrict__ mask) {
    size_t idx = (size_t)blockIdx.x * 256u + threadIdx.x;   // element index over N*N
    float b = bias[idx];
    unsigned int m = ballot32(b == 0.0f);
    if ((threadIdx.x & 31) == 0) mask[idx >> 5] = m;
}

// --------------------------------------------------------------------------
// K1: fts1 = x @ W1 (per head), bf16 WMMA, output packed as row-pairs:
//     fp1[h][m/2][d] = bf16(fts[m][d]) | bf16(fts[m+1][d])<<16
// --------------------------------------------------------------------------
__global__ __launch_bounds__(256) void k_gemm1(const float* __restrict__ x,
                                               const float* __restrict__ W1,
                                               unsigned int* __restrict__ fp1) {
    const int tid = threadIdx.x, wave = tid >> 5, lane = tid & 31;
    const int h = wave & 3;
    const int rowbase = blockIdx.x * 32 + (wave >> 2) * 16;
    const int hi = lane >> 4, c = lane & 15;
    const int row = rowbase + c;
    const float* xr = x + (size_t)row * kFIN;
    const float* Wh = W1 + (size_t)h * kFIN * kHID;

    v8f acc[4] = {};
    for (int ks = 0; ks < kFIN / 32; ++ks) {
        const int kb = ks * 32;
        BF16Frag A;
#pragma unroll
        for (int j = 0; j < 8; ++j) {                   // A 16x32: VGPR j = K{2j,2j+1}, lanes16-31 K+16
            int kA = 2 * j + hi * 16 + kb;
            float2 xv = *(const float2*)(xr + kA);
            A.w[j] = packbf(xv.x, xv.y);
        }
        BF16Frag B[4];
#pragma unroll
        for (int t = 0; t < 4; ++t) {
#pragma unroll
            for (int j = 0; j < 8; ++j) {               // B 32x16: VGPR j = K{2j,2j+1}, lanes16-31 K+16
                int kB = 2 * j + hi * 16 + kb;
                B[t].w[j] = packbf(Wh[(size_t)kB * kHID + t * 16 + c],
                                   Wh[(size_t)(kB + 1) * kHID + t * 16 + c]);
            }
        }
#pragma unroll
        for (int t = 0; t < 4; ++t)
            acc[t] = __builtin_amdgcn_wmma_f32_16x16x32_bf16(false, A.v, false, B[t].v,
                                                             (short)0, acc[t], false, false);
    }
    unsigned int* fph = fp1 + (size_t)h * (kN / 2) * kHID;
#pragma unroll
    for (int t = 0; t < 4; ++t)
#pragma unroll
        for (int e = 0; e < 8; e += 2) {                // C row = e + 8*hi
            int m = rowbase + e + hi * 8;
            fph[(size_t)(m >> 1) * kHID + t * 16 + c] = packbf(acc[t][e], acc[t][e + 1]);
        }
}

// --------------------------------------------------------------------------
// K2: f1/f2 attention vectors for layer 1 (from packed fts1).
// --------------------------------------------------------------------------
__global__ __launch_bounds__(256) void k_vecs1(const unsigned int* __restrict__ fp1,
                                               const float* __restrict__ a1s,
                                               const float* __restrict__ a1d,
                                               const float* __restrict__ b1s,
                                               const float* __restrict__ b1d,
                                               float* __restrict__ f1,
                                               float* __restrict__ f2) {
    int gid = blockIdx.x * 256 + threadIdx.x;           // (h, n)
    int h = gid >> 13, n = gid & (kN - 1);
    const unsigned int* r = fp1 + (size_t)h * (kN / 2) * kHID + (size_t)(n >> 1) * kHID;
    int sh = (n & 1) * 16;
    float s1 = 0.f, s2 = 0.f;
#pragma unroll 8
    for (int d = 0; d < kHID; ++d) {
        float v = bf2f((r[d] >> sh) & 0xFFFFu);
        s1 += v * a1s[h * kHID + d];
        s2 += v * a1d[h * kHID + d];
    }
    f1[gid] = s1 + b1s[h];
    f2[gid] = s2 + b1d[h];
}

// --------------------------------------------------------------------------
// K3/K7: per-row softmax stats (max, 1/sum) with masked rank-1 logits.
// One wave per (head, row). Shared by both layers.
// --------------------------------------------------------------------------
__global__ __launch_bounds__(256) void k_stats(const float* __restrict__ f1,
                                               const float* __restrict__ f2,
                                               const unsigned int* __restrict__ mask,
                                               float* __restrict__ rmax,
                                               float* __restrict__ rinv,
                                               int heads) {
    int wid = (blockIdx.x * 256 + threadIdx.x) >> 5;
    int lane = threadIdx.x & 31;
    int h = wid >> 13, n = wid & (kN - 1);
    if (h >= heads) return;
    const float f1r = f1[h * kN + n];
    const float* f2h = f2 + h * kN;
    const unsigned int* mrow = mask + (size_t)n * kMW;
    float m = -3.0e38f, s = 0.f;
    for (int ks = 0; ks < kMW; ++ks) {
        if (ks + 4 < kMW) __builtin_prefetch(mrow + ks + 4, 0, 0);
        unsigned int w = mrow[ks];
        float t = f1r + f2h[ks * 32 + lane];
        t = t > 0.f ? t : 0.2f * t;                      // leaky_relu(0.2)
        t += ((w >> lane) & 1u) ? 0.f : -1e9f;           // additive mask (matches reference)
        float nm = fmaxf(m, t);
        s = s * __expf(m - nm) + __expf(t - nm);
        m = nm;
    }
#pragma unroll
    for (int off = 16; off; off >>= 1) {                 // wave32 reduction
        float om = __shfl_xor(m, off, 32);
        float os = __shfl_xor(s, off, 32);
        float nm = fmaxf(m, om);
        s = s * __expf(m - nm) + os * __expf(om - nm);
        m = nm;
    }
    if (lane == 0) { rmax[h * kN + n] = m; rinv[h * kN + n] = 1.f / s; }
}

// --------------------------------------------------------------------------
// K4: layer-1 PV: vals = softmax(lrelu(f1+f2)+mask) @ fts1, ELU, head-concat.
// Block: 8 waves x 16 rows, one head (blockIdx.y). 256 k-steps of 32 columns.
// P fragment built in registers (lane owns row lane%16 -> stats loop-invariant).
// LDS tiles are double-buffered and filled with GLOBAL_LOAD_ASYNC_TO_LDS_B128,
// so the next tile's 4 KB is in flight while the current tile computes.
// --------------------------------------------------------------------------
__global__ __launch_bounds__(256) void k_pv1(const unsigned int* __restrict__ mask,
                                             const unsigned int* __restrict__ fp1,
                                             const float* __restrict__ f1,
                                             const float* __restrict__ f2,
                                             const float* __restrict__ rmax,
                                             const float* __restrict__ rinv,
                                             const float* __restrict__ bias1,
                                             unsigned short* __restrict__ h1bf) {
    __shared__ alignas(16) unsigned int pairs[2][16 * 64]; // [k/2][d] packed bf16 pairs
    __shared__ alignas(16) float f2s[2][32];
    const int tid = threadIdx.x, wave = tid >> 5, lane = tid & 31;
    const int hi = lane >> 4, c = lane & 15;
    const int h = blockIdx.y;
    const int rowbase = blockIdx.x * 128 + wave * 16;
    const int row = rowbase + c;
    const float f1r = f1[h * kN + row];
    const float mx  = rmax[h * kN + row];
    const float ri  = rinv[h * kN + row];
    const unsigned int* mrow = mask + (size_t)row * kMW;
    const unsigned int* fph = fp1 + (size_t)h * (kN / 2) * kHID;
    const float* f2h = f2 + h * kN;

    auto stage = [&](int ks, int buf) {
#if HAVE_ASYNC_LDS
        async_b128(fph + (size_t)ks * 1024 + tid * 4, &pairs[buf][tid * 4]);
        if (tid < 32) async_b32(f2h + ks * 32 + tid, &f2s[buf][tid]);
#else
        ((uint4*)pairs[buf])[tid] = ((const uint4*)(fph + (size_t)ks * 1024))[tid];
        if (tid < 32) f2s[buf][tid] = f2h[ks * 32 + tid];
#endif
    };

    stage(0, 0);
    v8f acc[4] = {};
    for (int ks = 0; ks < kMW; ++ks) {
        const int cur = ks & 1;
        if (ks + 1 < kMW) {
            stage(ks + 1, cur ^ 1);                      // next tile in flight
#if HAVE_ASYNC_LDS
            if (tid < 32) __builtin_amdgcn_s_wait_asynccnt(2);   // wave0 issued 2/tile
            else          __builtin_amdgcn_s_wait_asynccnt(1);
#endif
        }
#if HAVE_ASYNC_LDS
        else {
            __builtin_amdgcn_s_wait_asynccnt(0);
        }
#endif
        __syncthreads();                                 // current tile visible to all waves

        const unsigned int mw = mrow[ks];
        BF16Frag A;                                      // P tile, A 16x32 layout
#pragma unroll
        for (int j = 0; j < 8; ++j) {
            int e0 = 2 * j;
            int k0 = e0 + hi * 16;  // K index of elements 2j, 2j+1
            float t0 = f1r + f2s[cur][k0];
            float t1 = f1r + f2s[cur][k0 + 1];
            t0 = t0 > 0.f ? t0 : 0.2f * t0;
            t1 = t1 > 0.f ? t1 : 0.2f * t1;
            t0 += ((mw >> k0) & 1u) ? 0.f : -1e9f;
            t1 += ((mw >> (k0 + 1)) & 1u) ? 0.f : -1e9f;
            float p0 = __expf(t0 - mx) * ri;             // exp underflows to exact 0 off-edge
            float p1 = __expf(t1 - mx) * ri;
            A.w[j] = packbf(p0, p1);
        }
        BF16Frag B0, B1, B2, B3;
#pragma unroll
        for (int j = 0; j < 8; ++j) {
            int base = (j + hi * 8) * 64 + c;
            B0.w[j] = pairs[cur][base];
            B1.w[j] = pairs[cur][base + 16];
            B2.w[j] = pairs[cur][base + 32];
            B3.w[j] = pairs[cur][base + 48];
        }
        acc[0] = __builtin_amdgcn_wmma_f32_16x16x32_bf16(false, A.v, false, B0.v, (short)0, acc[0], false, false);
        acc[1] = __builtin_amdgcn_wmma_f32_16x16x32_bf16(false, A.v, false, B1.v, (short)0, acc[1], false, false);
        acc[2] = __builtin_amdgcn_wmma_f32_16x16x32_bf16(false, A.v, false, B2.v, (short)0, acc[2], false, false);
        acc[3] = __builtin_amdgcn_wmma_f32_16x16x32_bf16(false, A.v, false, B3.v, (short)0, acc[3], false, false);

        __syncthreads();                                 // all reads done before next overwrite
    }
#pragma unroll
    for (int t = 0; t < 4; ++t)
#pragma unroll
        for (int e = 0; e < 8; ++e) {
            int rr = rowbase + e + hi * 8;
            float v = acc[t][e] + bias1[h * kHID + t * 16 + c];
            v = v > 0.f ? v : __expf(v) - 1.f;           // ELU
            h1bf[(size_t)rr * (kNH * kHID) + h * kHID + t * 16 + c] = bf16bits(v);
        }
}

// --------------------------------------------------------------------------
// K5: fts2 = h1 @ W2, bf16 WMMA. Writes fp32 (for attn vecs) + packed pairs.
// --------------------------------------------------------------------------
__global__ __launch_bounds__(256) void k_gemm2(const unsigned short* __restrict__ h1bf,
                                               const float* __restrict__ W2,
                                               float* __restrict__ fts2,
                                               unsigned int* __restrict__ fp2) {
    const int tid = threadIdx.x, wave = tid >> 5, lane = tid & 31;
    const int hi = lane >> 4, c = lane & 15;
    const int rowbase = blockIdx.x * 128 + wave * 16;
    const int row = rowbase + c;
    const unsigned int* h1w = (const unsigned int*)h1bf;

    v8f acc = {};
    for (int ks = 0; ks < (kNH * kHID) / 32; ++ks) {
        const int kb = ks * 32;
        BF16Frag A, B;
#pragma unroll
        for (int j = 0; j < 8; ++j) {
            int kA = 2 * j + hi * 16 + kb;
            A.w[j] = h1w[(size_t)row * (kNH * kHID / 2) + (kA >> 1)];   // row-major -> pair dword
        }
#pragma unroll
        for (int j = 0; j < 8; ++j) {
            int kB = 2 * j + hi * 16 + kb;
            B.w[j] = packbf(W2[kB * kNCLS + c], W2[(kB + 1) * kNCLS + c]);
        }
        acc = __builtin_amdgcn_wmma_f32_16x16x32_bf16(false, A.v, false, B.v, (short)0, acc, false, false);
    }
#pragma unroll
    for (int e = 0; e < 8; ++e) {
        int rr = rowbase + e + hi * 8;
        fts2[(size_t)rr * kNCLS + c] = acc[e];
    }
#pragma unroll
    for (int e = 0; e < 8; e += 2) {
        int m = rowbase + e + hi * 8;
        fp2[(size_t)(m >> 1) * kNCLS + c] = packbf(acc[e], acc[e + 1]);
    }
}

// --------------------------------------------------------------------------
// K6: f1/f2 vectors for layer 2.
// --------------------------------------------------------------------------
__global__ __launch_bounds__(256) void k_vecs2(const float* __restrict__ fts2,
                                               const float* __restrict__ a2s,
                                               const float* __restrict__ a2d,
                                               const float* __restrict__ b2s,
                                               const float* __restrict__ b2d,
                                               float* __restrict__ f1,
                                               float* __restrict__ f2) {
    int n = blockIdx.x * 256 + threadIdx.x;
    float s1 = 0.f, s2 = 0.f;
#pragma unroll
    for (int c2 = 0; c2 < kNCLS; ++c2) {
        float v = fts2[(size_t)n * kNCLS + c2];
        s1 += v * a2s[c2];
        s2 += v * a2d[c2];
    }
    f1[n] = s1 + b2s[0];
    f2[n] = s2 + b2d[0];
}

// --------------------------------------------------------------------------
// K8: layer-2 PV: out = softmax(...) @ fts2 + bias2 (identity activation).
// --------------------------------------------------------------------------
__global__ __launch_bounds__(256) void k_pv2(const unsigned int* __restrict__ mask,
                                             const unsigned int* __restrict__ fp2,
                                             const float* __restrict__ f1,
                                             const float* __restrict__ f2,
                                             const float* __restrict__ rmax,
                                             const float* __restrict__ rinv,
                                             const float* __restrict__ bias2,
                                             float* __restrict__ out) {
    __shared__ alignas(16) unsigned int pairs[2][16 * 16];
    __shared__ alignas(16) float f2s[2][32];
    const int tid = threadIdx.x, wave = tid >> 5, lane = tid & 31;
    const int hi = lane >> 4, c = lane & 15;
    const int rowbase = blockIdx.x * 128 + wave * 16;
    const int row = rowbase + c;
    const float f1r = f1[row], mx = rmax[row], ri = rinv[row];
    const unsigned int* mrow = mask + (size_t)row * kMW;

    auto stage = [&](int ks, int buf) {
#if HAVE_ASYNC_LDS
        if (tid < 64) async_b128(fp2 + (size_t)ks * 256 + tid * 4, &pairs[buf][tid * 4]);
        if (tid < 32) async_b32(f2 + ks * 32 + tid, &f2s[buf][tid]);
#else
        if (tid < 64) ((uint4*)pairs[buf])[tid] = ((const uint4*)(fp2 + (size_t)ks * 256))[tid];
        if (tid < 32) f2s[buf][tid] = f2[ks * 32 + tid];
#endif
    };

    stage(0, 0);
    v8f acc = {};
    for (int ks = 0; ks < kMW; ++ks) {
        const int cur = ks & 1;
        if (ks + 1 < kMW) {
            stage(ks + 1, cur ^ 1);
#if HAVE_ASYNC_LDS
            if (tid < 32)      __builtin_amdgcn_s_wait_asynccnt(2);  // wave0: b128+b32 per tile
            else if (tid < 64) __builtin_amdgcn_s_wait_asynccnt(1);  // wave1: b128 per tile
            else               __builtin_amdgcn_s_wait_asynccnt(0);
#endif
        }
#if HAVE_ASYNC_LDS
        else {
            __builtin_amdgcn_s_wait_asynccnt(0);
        }
#endif
        __syncthreads();

        const unsigned int mw = mrow[ks];
        BF16Frag A;
#pragma unroll
        for (int j = 0; j < 8; ++j) {
            int e0 = 2 * j;
            int k0 = e0 + hi * 16;
            float t0 = f1r + f2s[cur][k0];
            float t1 = f1r + f2s[cur][k0 + 1];
            t0 = t0 > 0.f ? t0 : 0.2f * t0;
            t1 = t1 > 0.f ? t1 : 0.2f * t1;
            t0 += ((mw >> k0) & 1u) ? 0.f : -1e9f;
            t1 += ((mw >> (k0 + 1)) & 1u) ? 0.f : -1e9f;
            A.w[j] = packbf(__expf(t0 - mx) * ri, __expf(t1 - mx) * ri);
        }
        BF16Frag B;
#pragma unroll
        for (int j = 0; j < 8; ++j)
            B.w[j] = pairs[cur][(j + hi * 8) * 16 + c];
        acc = __builtin_amdgcn_wmma_f32_16x16x32_bf16(false, A.v, false, B.v, (short)0, acc, false, false);

        __syncthreads();
    }
#pragma unroll
    for (int e = 0; e < 8; ++e) {
        int rr = rowbase + e + hi * 8;
        out[(size_t)rr * kNCLS + c] = acc[e] + bias2[c];
    }
}

// --------------------------------------------------------------------------
extern "C" void kernel_launch(void* const* d_in, const int* in_sizes, int n_in,
                              void* d_out, int out_size, void* d_ws, size_t ws_size,
                              hipStream_t stream) {
    (void)in_sizes; (void)n_in; (void)out_size; (void)ws_size;
    const float* seq   = (const float*)d_in[0];
    const float* biasM = (const float*)d_in[1];
    const float* W1    = (const float*)d_in[2];
    const float* a1s   = (const float*)d_in[3];
    const float* a1d   = (const float*)d_in[4];
    const float* b1s   = (const float*)d_in[5];
    const float* b1d   = (const float*)d_in[6];
    const float* bias1 = (const float*)d_in[7];
    const float* W2    = (const float*)d_in[8];
    const float* a2s   = (const float*)d_in[9];
    const float* a2d   = (const float*)d_in[10];
    const float* b2s   = (const float*)d_in[11];
    const float* b2d   = (const float*)d_in[12];
    const float* bias2 = (const float*)d_in[13];
    float* out = (float*)d_out;

    char* ws = (char*)d_ws;
    unsigned int*  mask  = (unsigned int*)(ws);                       //  8 MB bitset
    unsigned int*  fp1   = (unsigned int*)(ws + 8388608);             //  4 MB fts1 pairs
    float*         f1_1  = (float*)(ws + 12582912);                   // 128 KB
    float*         f2_1  = (float*)(ws + 12713984);
    float*         rmax1 = (float*)(ws + 12845056);
    float*         rinv1 = (float*)(ws + 12976128);
    unsigned short* h1bf = (unsigned short*)(ws + 13107200);          //  4 MB
    float*         fts2  = (float*)(ws + 17301504);                   // 512 KB
    unsigned int*  fp2   = (unsigned int*)(ws + 17825792);            // 256 KB
    float*         f1_2  = (float*)(ws + 18087936);
    float*         f2_2  = (float*)(ws + 18120704);
    float*         rmax2 = (float*)(ws + 18153472);
    float*         rinv2 = (float*)(ws + 18186240);

    k_build_mask<<<dim3((unsigned)((size_t)kN * kN / 256)), 256, 0, stream>>>(biasM, mask);
    k_gemm1<<<dim3(kN / 32), 256, 0, stream>>>(seq, W1, fp1);
    k_vecs1<<<dim3(kNH * kN / 256), 256, 0, stream>>>(fp1, a1s, a1d, b1s, b1d, f1_1, f2_1);
    k_stats<<<dim3(kNH * kN / 8), 256, 0, stream>>>(f1_1, f2_1, mask, rmax1, rinv1, kNH);
    k_pv1<<<dim3(kN / 128, kNH), 256, 0, stream>>>(mask, fp1, f1_1, f2_1, rmax1, rinv1, bias1, h1bf);
    k_gemm2<<<dim3(kN / 128), 256, 0, stream>>>(h1bf, W2, fts2, fp2);
    k_vecs2<<<dim3(kN / 256), 256, 0, stream>>>(fts2, a2s, a2d, b2s, b2d, f1_2, f2_2);
    k_stats<<<dim3(kN / 8), 256, 0, stream>>>(f1_2, f2_2, mask, rmax2, rinv2, 1);
    k_pv2<<<dim3(kN / 128), 256, 0, stream>>>(mask, fp2, f1_2, f2_2, rmax2, rinv2, bias2, out);
}
